// GCNPolicy_31353261261344
// MI455X (gfx1250) — matrix operand order repaired and checked by reading, not compiled, not run on hardware
//
#include <hip/hip_runtime.h>
#include <hip/hip_bf16.h>

// ---------------------------------------------------------------------------
// Bipartite GCN policy net on MI455X (gfx1250, wave32, WMMA).
//
// Dominant cost: per-edge 64x64 matmul (2 convs x 1M edges x 4096 MAC =
// 16.4 GFLOP) plus 128M tanh evaluations.  All node arrays (<= 25.6 MB) fit
// in the 192 MB L2, so the edge stage is compute-bound -> f16 WMMA
// (v_wmma_f32_16x16x32_f16, f32 accumulate) + hardware v_tanh_f32.
// Per-node terms right@Wl and left@Wr are hoisted out of the edge loop,
// then each edge does
//   msg = tanh((tl[tgt] + tr[src] + ef*We) * pre_scale) @ Wf + bf
// and scatter-adds msg into agg[tgt] with global_atomic_add_f32.
// ---------------------------------------------------------------------------

typedef __attribute__((ext_vector_type(16))) _Float16 v16h;
typedef __attribute__((ext_vector_type(8)))  _Float16 v8h;
typedef __attribute__((ext_vector_type(4)))  _Float16 v4h;
typedef __attribute__((ext_vector_type(8)))  float    v8f;

#define NCONS 50000
#define NVARS 100000
#define NEDGE 1000000
#define NB    8

// gfx1250 has a hardware tanh (V_TANH_F32, TRANS32 pipe).  Use it if the
// toolchain declares the builtin, else fall back to libm.
#if defined(__gfx1250__) && __has_builtin(__builtin_amdgcn_tanhf)
static __device__ __forceinline__ float fast_tanh(float x) { return __builtin_amdgcn_tanhf(x); }
#else
static __device__ __forceinline__ float fast_tanh(float x) { return tanhf(x); }
#endif

static __device__ __forceinline__ v8f wmma16(const v16h a, const v16h b, const v8f c) {
  return __builtin_amdgcn_wmma_f32_16x16x32_f16(false, a, false, b, (short)0, c, false, false);
}

// ---------------------------------------------------------------------------
// First embedding layer: out[n,64] = tanh( ((x + shift)*scale) @ W1 + b1 )
// nin is tiny (5 or 17) -> plain VALU, memory-light.
// ---------------------------------------------------------------------------
__global__ __launch_bounds__(256) void emb_first_kernel(
    const float* __restrict__ x, int nin,
    const float* __restrict__ shift, const float* __restrict__ scale,
    const float* __restrict__ W1, const float* __restrict__ b1,
    float* __restrict__ out, int n) {
  __shared__ float sW[17 * 64];
  __shared__ float ssh[17], ssc[17], sb[64];
  for (int i = threadIdx.x; i < nin * 64; i += 256) sW[i] = W1[i];
  if (threadIdx.x < nin) { ssh[threadIdx.x] = shift[threadIdx.x]; ssc[threadIdx.x] = scale[threadIdx.x]; }
  if (threadIdx.x < 64) sb[threadIdx.x] = b1[threadIdx.x];
  __syncthreads();
  size_t total = (size_t)n * 64;
  for (size_t idx = (size_t)blockIdx.x * 256 + threadIdx.x; idx < total;
       idx += (size_t)gridDim.x * 256) {
    int r = (int)(idx >> 6), j = (int)(idx & 63);
    float a = sb[j];
    for (int k = 0; k < nin; k++)
      a += (x[(size_t)r * nin + k] + ssh[k]) * ssc[k] * sW[k * 64 + j];
    out[idx] = fast_tanh(a);
  }
}

// ---------------------------------------------------------------------------
// Generic node MLP layer via WMMA:
//   out[r,:] = act( [in0[r]*s0 (, in1[r])] @ W[K,64] + bias )
// KCH = number of 32-wide K chunks (2 -> K=64, 4 -> K=128 concat form).
// Each wave owns a 16-row tile; W lives in VGPRs in WMMA B-layout.
// ---------------------------------------------------------------------------
template <int KCH>
__global__ __launch_bounds__(256) void mlp64_wmma_kernel(
    const float* __restrict__ in0, const float* __restrict__ scale0,
    const float* __restrict__ in1,
    const float* __restrict__ W, const float* __restrict__ bias,
    float* __restrict__ out, int nrows, int act_tanh) {
  constexpr int K = KCH * 32;
  __shared__ float    sW[K * 64];
  __shared__ _Float16 sA[8][16 * K];
  const int tid = threadIdx.x;
  for (int i = tid; i < K * 64; i += 256) sW[i] = W[i];
  __syncthreads();
  const int lane = tid & 31, w = tid >> 5;
  const int col = lane & 15, hi = lane >> 4;
  const int cidx = 4 * col;   // this lane's 4-column slice when staging rows

  // B-matrix (K x 64) resident in VGPRs, contiguous k order (same as A):
  // lanes 0-15 hold K=c*32+0..15 of column `col`; lanes 16-31 hold K=c*32+16..31.
  v16h Bv[KCH][4];
#pragma unroll
  for (int c = 0; c < KCH; c++)
#pragma unroll
    for (int t = 0; t < 4; t++)
#pragma unroll
      for (int j = 0; j < 16; j++)
        Bv[c][t][j] = (_Float16)sW[(c * 32 + hi * 16 + j) * 64 + t * 16 + col];

  float bv[4];
#pragma unroll
  for (int t = 0; t < 4; t++) bv[t] = bias ? bias[t * 16 + col] : 0.0f;
  const float s0 = scale0 ? *scale0 : 1.0f;

  const int ngroups = (nrows + 15) >> 4;
  for (int g = blockIdx.x * 8 + w; g < ngroups; g += gridDim.x * 8) {
    const int base = g << 4;
    // Stage 16 rows of A (f32 -> f16) into this wave's LDS slab, row-major.
    // Half-wave hi covers row 2*it+hi, 4 columns per lane -> b128 loads.
#pragma unroll 1
    for (int it = 0; it < 8; it++) {
      const int e = 2 * it + hi;
      const int r = base + e;
      float4 a0 = make_float4(0.f, 0.f, 0.f, 0.f);
      if (r < nrows) a0 = *(const float4*)&in0[(size_t)r * 64 + cidx];
      v4h h0; h0[0] = (_Float16)(a0.x * s0); h0[1] = (_Float16)(a0.y * s0);
      h0[2] = (_Float16)(a0.z * s0); h0[3] = (_Float16)(a0.w * s0);
      *(v4h*)&sA[w][e * K + cidx] = h0;
      if constexpr (KCH == 4) {
        float4 a1 = make_float4(0.f, 0.f, 0.f, 0.f);
        if (r < nrows) a1 = *(const float4*)&in1[(size_t)r * 64 + cidx];
        v4h h1; h1[0] = (_Float16)a1.x; h1[1] = (_Float16)a1.y;
        h1[2] = (_Float16)a1.z; h1[3] = (_Float16)a1.w;
        *(v4h*)&sA[w][e * K + 64 + cidx] = h1;
      }
    }
    v8f acc[4];
#pragma unroll
    for (int t = 0; t < 4; t++)
#pragma unroll
      for (int r2 = 0; r2 < 8; r2++) acc[t][r2] = bv[t];

#pragma unroll
    for (int c = 0; c < KCH; c++) {
      // A-layout: lane `row` (=col) elems 0..15 = K hi*16+0..15
      const _Float16* rp = &sA[w][col * K + c * 32 + hi * 16];
      const v8h lo = *(const v8h*)rp;
      const v8h hh = *(const v8h*)(rp + 8);
      v16h a;
#pragma unroll
      for (int j = 0; j < 8; j++) { a[j] = lo[j]; a[8 + j] = hh[j]; }
#pragma unroll
      for (int t = 0; t < 4; t++) acc[t] = wmma16(a, Bv[c][t], acc[t]);
    }
    // C/D layout: VGPR r2 holds row r2 + 8*hi, column t*16+col.
#pragma unroll
    for (int t = 0; t < 4; t++)
#pragma unroll
      for (int r2 = 0; r2 < 8; r2++) {
        const int r = base + r2 + hi * 8;
        if (r < nrows) {
          const float v = acc[t][r2];
          out[(size_t)r * 64 + t * 16 + col] = act_tanh ? fast_tanh(v) : v;
        }
      }
  }
}

// ---------------------------------------------------------------------------
// Edge convolution core (the 16.4 GFLOP hot loop):
// per wave: 16 edges -> joint rows staged in LDS (f16, WMMA A-layout reads),
// Wf resident in VGPRs (B-layout), 8 WMMAs, f32 atomic scatter into agg[tgt].
// ---------------------------------------------------------------------------
__global__ __launch_bounds__(256) void edge_conv_wmma_kernel(
    const int* __restrict__ tgt, const int* __restrict__ src,
    const float* __restrict__ ef, const float* __restrict__ e_shift,
    const float* __restrict__ e_scale,
    const float* __restrict__ tl, const float* __restrict__ tr,
    const float* __restrict__ We, const float* __restrict__ Wf,
    const float* __restrict__ bf, const float* __restrict__ pre_scale,
    float* __restrict__ agg, int ne) {
  __shared__ float    sWf[64 * 64];
  __shared__ _Float16 sA[8][16 * 64];
  __shared__ int      sTgt[8][16];
  __shared__ int      sSrc[8][16];
  __shared__ float    sEf[8][16];
  const int tid = threadIdx.x;
  for (int i = tid; i < 4096; i += 256) sWf[i] = Wf[i];
  __syncthreads();
  const int lane = tid & 31, w = tid >> 5;
  const int col = lane & 15, hi = lane >> 4;
  const int cidx = 4 * col;

  v16h Bv[2][4];
#pragma unroll
  for (int c = 0; c < 2; c++)
#pragma unroll
    for (int t = 0; t < 4; t++)
#pragma unroll
      for (int j = 0; j < 16; j++)
        Bv[c][t][j] = (_Float16)sWf[(c * 32 + hi * 16 + j) * 64 + t * 16 + col];

  float bv[4];
#pragma unroll
  for (int t = 0; t < 4; t++) bv[t] = bf[t * 16 + col];
  const float esh = *e_shift, esc = *e_scale, ps = *pre_scale;
  const float4 we4 = *(const float4*)&We[cidx];

  const int ngroups = (ne + 15) >> 4;
  const int gstride = gridDim.x * 8;
  for (int g = blockIdx.x * 8 + w; g < ngroups; g += gstride) {
    const int base = g << 4;
    if (lane < 16) {
      const int e  = base + lane;
      const int ec = e < ne ? e : ne - 1;
      sTgt[w][lane] = tgt[ec];
      sSrc[w][lane] = src[ec];
      sEf[w][lane]  = (ef[ec] + esh) * esc;
      const int gn = g + gstride;   // prefetch next group's edge stream
      if (gn < ngroups) {
        __builtin_prefetch(&tgt[(gn << 4) + lane], 0, 1);
        __builtin_prefetch(&src[(gn << 4) + lane], 0, 1);
      }
    }
    // joint = tl[tgt] + tr[src] + ef*We ; A-tile = tanh(joint * pre_scale)
    // Half-wave hi covers edge 2*it+hi, 4 columns per lane -> b128 gathers.
#pragma unroll 1
    for (int it = 0; it < 8; it++) {
      const int e16 = 2 * it + hi;
      const int tg = sTgt[w][e16], sr = sSrc[w][e16];
      const float ev = sEf[w][e16];
      const float4 a = *(const float4*)&tl[(size_t)tg * 64 + cidx];
      const float4 b = *(const float4*)&tr[(size_t)sr * 64 + cidx];
      float t0 = fast_tanh((a.x + b.x + ev * we4.x) * ps);
      float t1 = fast_tanh((a.y + b.y + ev * we4.y) * ps);
      float t2 = fast_tanh((a.z + b.z + ev * we4.z) * ps);
      float t3 = fast_tanh((a.w + b.w + ev * we4.w) * ps);
      if (base + e16 >= ne) { t0 = t1 = t2 = t3 = 0.f; }
      v4h hv; hv[0] = (_Float16)t0; hv[1] = (_Float16)t1;
      hv[2] = (_Float16)t2; hv[3] = (_Float16)t3;
      *(v4h*)&sA[w][e16 * 64 + cidx] = hv;
    }
    v8f acc[4];
#pragma unroll
    for (int t = 0; t < 4; t++)
#pragma unroll
      for (int r2 = 0; r2 < 8; r2++) acc[t][r2] = bv[t];   // bf added per edge
#pragma unroll
    for (int c = 0; c < 2; c++) {
      const _Float16* rp = &sA[w][col * 64 + c * 32 + hi * 16];
      const v8h lo = *(const v8h*)rp;
      const v8h hh = *(const v8h*)(rp + 8);
      v16h a;
#pragma unroll
      for (int j = 0; j < 8; j++) { a[j] = lo[j]; a[8 + j] = hh[j]; }
#pragma unroll
      for (int t = 0; t < 4; t++) acc[t] = wmma16(a, Bv[c][t], acc[t]);
    }
    // scatter-add msg into agg[tgt]
#pragma unroll
    for (int t = 0; t < 4; t++)
#pragma unroll
      for (int r2 = 0; r2 < 8; r2++) {
        const int m = r2 + hi * 8;
        if (base + m < ne) {
          const int tg = sTgt[w][m];
          atomicAdd(&agg[(size_t)tg * 64 + t * 16 + col], acc[t][r2]);
        }
      }
  }
}

// ---------------------------------------------------------------------------
// Per-sample mean pooling over variable nodes.
// ---------------------------------------------------------------------------
__global__ __launch_bounds__(256) void pool_kernel(
    const float* __restrict__ var2, const int* __restrict__ nvps,
    float* __restrict__ pooled) {
  const int b = blockIdx.x;
  int start = 0;
  for (int i = 0; i < b; i++) start += nvps[i];
  const int cnt = nvps[b];
  const int col = threadIdx.x & 63, sub = threadIdx.x >> 6;
  float s = 0.f;
  for (int r = sub; r < cnt; r += 4) s += var2[(size_t)(start + r) * 64 + col];
  __shared__ float red[256];
  red[threadIdx.x] = s;
  __syncthreads();
  if (sub == 0) {
    const float tot = red[col] + red[64 + col] + red[128 + col] + red[192 + col];
    pooled[b * 64 + col] = tot / (float)cnt;
  }
}

// ---------------------------------------------------------------------------
// Heads (B=8, tiny): o1 = head64->14, pc = [pooled,o1], o2/o3 = head78->56.
// ---------------------------------------------------------------------------
__global__ __launch_bounds__(256) void heads_kernel(
    const float* __restrict__ pooled,
    const float* f1W1, const float* f1b1, const float* f1W2, const float* f1b2,
    const float* f2W1, const float* f2b1, const float* f2W2, const float* f2b2,
    const float* f3W1, const float* f3b1, const float* f3W2, const float* f3b2,
    float* __restrict__ out) {
  __shared__ float sp[8 * 64];
  __shared__ float pc[8 * 78];
  __shared__ float h[8 * 64];
  const int tid = threadIdx.x;
  for (int i = tid; i < 512; i += 256) {
    sp[i] = pooled[i];
    pc[(i >> 6) * 78 + (i & 63)] = pooled[i];
  }
  __syncthreads();
  for (int i = tid; i < 512; i += 256) {
    const int r = i >> 6, j = i & 63;
    float a = f1b1[j];
    for (int k = 0; k < 64; k++) a += fast_tanh(sp[r * 64 + k]) * f1W1[k * 64 + j];
    h[i] = fast_tanh(a);
  }
  __syncthreads();
  for (int i = tid; i < 8 * 14; i += 256) {
    const int r = i / 14, j = i % 14;
    float a = f1b2[j];
    for (int k = 0; k < 64; k++) a += h[r * 64 + k] * f1W2[k * 14 + j];
    out[i] = a;
    pc[r * 78 + 64 + j] = a;
  }
  __syncthreads();
  for (int i = tid; i < 512; i += 256) {
    const int r = i >> 6, j = i & 63;
    float a = f2b1[j];
    for (int k = 0; k < 78; k++) a += fast_tanh(pc[r * 78 + k]) * f2W1[k * 64 + j];
    h[i] = fast_tanh(a);
  }
  __syncthreads();
  for (int i = tid; i < 8 * 56; i += 256) {
    const int r = i / 56, j = i % 56;
    float a = f2b2[j];
    for (int k = 0; k < 64; k++) a += h[r * 64 + k] * f2W2[k * 56 + j];
    out[112 + i] = a;
  }
  __syncthreads();
  for (int i = tid; i < 512; i += 256) {
    const int r = i >> 6, j = i & 63;
    float a = f3b1[j];
    for (int k = 0; k < 78; k++) a += fast_tanh(pc[r * 78 + k]) * f3W1[k * 64 + j];
    h[i] = fast_tanh(a);
  }
  __syncthreads();
  for (int i = tid; i < 8 * 56; i += 256) {
    const int r = i / 56, j = i % 56;
    float a = f3b2[j];
    for (int k = 0; k < 64; k++) a += h[r * 64 + k] * f3W2[k * 56 + j];
    out[560 + i] = a;
  }
}

// ---------------------------------------------------------------------------
// Host orchestration.  Input order: setup_inputs() insertion order with the
// nested `params` dict flattened in source order.
// ---------------------------------------------------------------------------
extern "C" void kernel_launch(void* const* d_in, const int* in_sizes, int n_in,
                              void* d_out, int out_size, void* d_ws, size_t ws_size,
                              hipStream_t stream) {
  (void)in_sizes; (void)n_in; (void)out_size; (void)ws_size;
  const float* cons_x = (const float*)d_in[0];
  const int*   ci     = (const int*)d_in[1];             // edge_indices[0]
  const int*   vi     = ((const int*)d_in[1]) + NEDGE;   // edge_indices[1]
  const float* efeat  = (const float*)d_in[2];
  const float* var_x  = (const float*)d_in[3];
  const int*   nvps   = (const int*)d_in[5];

  // params flattened (insertion order):
  // 6..11  cons_emb: shift, scale, W1, b1, W2, b2
  // 12..17 var_emb : shift, scale, W1, b1, W2, b2
  // 18..19 edge_emb: shift, scale
  // 20..31 conv_v_to_c: Wl,bl,We,Wr,pre,Wf,bf,post,Wo1,bo1,Wo2,bo2
  // 32..43 conv_c_to_v: same
  // 44..47 fc1: W1,b1,W2,b2 ; 48..51 fc2 ; 52..55 fc3
  const float* P[56];
  for (int i = 0; i < 56; i++) P[i] = (const float*)d_in[i];

  float* ws     = (float*)d_ws;
  float* consE  = ws;                                  // [NC,64]
  float* varE   = consE + (size_t)NCONS * 64;          // [NV,64]
  float* bTL    = varE  + (size_t)NVARS * 64;          // [<=NV,64]
  float* bTR    = bTL   + (size_t)NVARS * 64;          // [<=NV,64]
  float* bAGG   = bTR   + (size_t)NVARS * 64;          // [<=NV,64]
  float* bTMP   = bAGG  + (size_t)NVARS * 64;          // [<=NV,64]
  float* cons2  = bTMP  + (size_t)NVARS * 64;          // [NC,64]
  float* pooled = cons2 + (size_t)NCONS * 64;          // [8,64]

  auto mlpBlocks = [](int n) { int g = (n + 15) / 16; int b = (g + 7) / 8; return b > 1024 ? 1024 : b; };

  // ---- embeddings ----
  emb_first_kernel<<<4096, 256, 0, stream>>>(cons_x, 5, P[6], P[7], P[8], P[9], bTMP, NCONS);
  mlp64_wmma_kernel<2><<<mlpBlocks(NCONS), 256, 0, stream>>>(bTMP, nullptr, nullptr, P[10], P[11], consE, NCONS, 1);
  emb_first_kernel<<<4096, 256, 0, stream>>>(var_x, 17, P[12], P[13], P[14], P[15], bTMP, NVARS);
  mlp64_wmma_kernel<2><<<mlpBlocks(NVARS), 256, 0, stream>>>(bTMP, nullptr, nullptr, P[16], P[17], varE, NVARS, 1);

  // ---- conv v->c : tgt = cons, src = var ----
  mlp64_wmma_kernel<2><<<mlpBlocks(NCONS), 256, 0, stream>>>(consE, nullptr, nullptr, P[20], P[21], bTL, NCONS, 0); // tl = cons@Wl+bl
  mlp64_wmma_kernel<2><<<mlpBlocks(NVARS), 256, 0, stream>>>(varE,  nullptr, nullptr, P[23], nullptr, bTR, NVARS, 0); // tr = var@Wr
  hipMemsetAsync(bAGG, 0, (size_t)NCONS * 64 * sizeof(float), stream);
  edge_conv_wmma_kernel<<<1024, 256, 0, stream>>>(ci, vi, efeat, P[18], P[19], bTL, bTR,
                                                  P[22], P[25], P[26], P[24], bAGG, NEDGE);
  mlp64_wmma_kernel<4><<<mlpBlocks(NCONS), 256, 0, stream>>>(bAGG, P[27], consE, P[28], P[29], bTMP, NCONS, 1); // tanh([agg*post,cons]@Wo1+bo1)
  mlp64_wmma_kernel<2><<<mlpBlocks(NCONS), 256, 0, stream>>>(bTMP, nullptr, nullptr, P[30], P[31], cons2, NCONS, 0);

  // ---- conv c->v : tgt = var, src = cons ----
  mlp64_wmma_kernel<2><<<mlpBlocks(NVARS), 256, 0, stream>>>(varE,  nullptr, nullptr, P[32], P[33], bTL, NVARS, 0); // tl = var@Wl+bl
  mlp64_wmma_kernel<2><<<mlpBlocks(NCONS), 256, 0, stream>>>(cons2, nullptr, nullptr, P[35], nullptr, bTR, NCONS, 0); // tr = cons2@Wr
  hipMemsetAsync(bAGG, 0, (size_t)NVARS * 64 * sizeof(float), stream);
  edge_conv_wmma_kernel<<<1024, 256, 0, stream>>>(vi, ci, efeat, P[18], P[19], bTL, bTR,
                                                  P[34], P[37], P[38], P[36], bAGG, NEDGE);
  mlp64_wmma_kernel<4><<<mlpBlocks(NVARS), 256, 0, stream>>>(bAGG, P[39], varE, P[40], P[41], bTMP, NVARS, 1);
  mlp64_wmma_kernel<2><<<mlpBlocks(NVARS), 256, 0, stream>>>(bTMP, nullptr, nullptr, P[42], P[43], bTL, NVARS, 0); // var2 -> bTL

  // ---- pooling + heads ----
  pool_kernel<<<NB, 256, 0, stream>>>(bTL, nvps, pooled);
  heads_kernel<<<1, 256, 0, stream>>>(pooled,
      P[44], P[45], P[46], P[47],
      P[48], P[49], P[50], P[51],
      P[52], P[53], P[54], P[55],
      (float*)d_out);
}
